// BasicSelfAttention2D_48326972015259
// MI455X (gfx1250) — matrix operand order repaired and checked by reading, not compiled, not run on hardware
//
#include <hip/hip_runtime.h>

typedef __attribute__((ext_vector_type(16))) __bf16 v16bf;
typedef __attribute__((ext_vector_type(8)))  float  v8f;
typedef __attribute__((ext_vector_type(4)))  unsigned v4u;
typedef __attribute__((ext_vector_type(8)))  int      v8i;
typedef __attribute__((ext_vector_type(4)))  int      v4i;

union Frag16 { v16bf v; unsigned u[8]; };

#define DIMB 8
#define DIMC 256
#define DIMH 64
#define DIMW 64
#define DIMN 4096   // H*W
#define DIMNK 1024  // (H/2)*(W/2)
#define DIMD 32     // C/8

#if defined(__HIP_DEVICE_COMPILE__) && __has_builtin(__builtin_amdgcn_tensor_load_to_lds) && __has_builtin(__builtin_amdgcn_s_wait_tensorcnt)
#define USE_TDM 1
#else
#define USE_TDM 0
#endif

// ---------------------------------------------------------------------------
// Elementwise prep kernels
// ---------------------------------------------------------------------------
__global__ void cvt_f32_bf16_kernel(const float* __restrict__ src,
                                    __bf16* __restrict__ dst, int n) {
  int i = blockIdx.x * blockDim.x + threadIdx.x;
  if (i < n) dst[i] = (__bf16)src[i];
}

// x [B,C,H,W] f32  ->  x_t [B,N,C] bf16
__global__ void transpose_x_kernel(const float* __restrict__ x,
                                   __bf16* __restrict__ xt) {
  size_t i = (size_t)blockIdx.x * blockDim.x + threadIdx.x;
  size_t total = (size_t)DIMB * DIMN * DIMC;
  if (i >= total) return;
  int c = (int)(i % DIMC);
  size_t t = i / DIMC;
  int n = (int)(t % DIMN);
  int b = (int)(t / DIMN);
  xt[i] = (__bf16)x[((size_t)b * DIMC + c) * DIMN + n];
}

// avgpool2x2 -> kvc [B,C,Nk] and kvt [B,Nk,C]
__global__ void avgpool_kernel(const float* __restrict__ x,
                               __bf16* __restrict__ kvc,
                               __bf16* __restrict__ kvt) {
  size_t i = (size_t)blockIdx.x * blockDim.x + threadIdx.x;
  size_t total = (size_t)DIMB * DIMC * DIMNK;
  if (i >= total) return;
  int m = (int)(i % DIMNK);
  size_t t = i / DIMNK;
  int c = (int)(t % DIMC);
  int b = (int)(t / DIMC);
  int hk = m / (DIMW / 2), wk = m % (DIMW / 2);
  const float* p = x + (((size_t)b * DIMC + c) * DIMH + hk * 2) * DIMW + wk * 2;
  float v = 0.25f * (p[0] + p[1] + p[DIMW] + p[DIMW + 1]);
  kvc[((size_t)b * DIMC + c) * DIMNK + m] = (__bf16)v;
  kvt[((size_t)b * DIMNK + m) * DIMC + c] = (__bf16)v;
}

// out[b,c,n] = x[b,c,n] + gamma * y_t[b,n,c]
__global__ void final_add_kernel(const float* __restrict__ x,
                                 const float* __restrict__ yt,
                                 const float* __restrict__ gamma,
                                 float* __restrict__ out) {
  size_t i = (size_t)blockIdx.x * blockDim.x + threadIdx.x;
  size_t total = (size_t)DIMB * DIMC * DIMN;
  if (i >= total) return;
  int n = (int)(i % DIMN);
  size_t t = i / DIMN;
  int c = (int)(t % DIMC);
  int b = (int)(t / DIMC);
  out[i] = x[i] + gamma[0] * yt[((size_t)b * DIMN + n) * DIMC + c];
}

// ---------------------------------------------------------------------------
// Generic WMMA GEMM: D[m,n] = sum_k A[m,k] * (bTrans ? B[n,k] : B[k,n])
// One 16x16 tile per wave; bf16 16x16x32 WMMA; K-pairs loaded as dwords.
// ---------------------------------------------------------------------------
template <typename OutT>
__global__ void wmma_gemm_kernel(const __bf16* __restrict__ A, long long sAb, int lda,
                                 const __bf16* __restrict__ Bm, long long sBb, int ldb,
                                 int bTrans,
                                 OutT* __restrict__ D, long long sDb, int ldo,
                                 int mt, int nt, int nb, int K) {
  int wave = threadIdx.x >> 5;
  int lane = threadIdx.x & 31;
  int tile = blockIdx.x * (blockDim.x >> 5) + wave;
  int total = mt * nt * nb;
  if (tile >= total) return;            // wave-uniform: EXEC stays all-ones
  int bidx = tile / (mt * nt);
  int rem  = tile % (mt * nt);
  int tm = rem / nt, tn = rem % nt;

  const __bf16* Ab = A + (size_t)bidx * sAb;
  const __bf16* Bb = Bm + (size_t)bidx * sBb;
  OutT*         Db = D + (size_t)bidx * sDb;

  int half = lane >> 4, l16 = lane & 15;
  const __bf16* arow = Ab + (size_t)(tm * 16 + l16) * lda;
  int ncol = tn * 16 + l16;
  const __bf16* brow = Bb + (size_t)ncol * ldb;   // used when bTrans

  v8f acc = {};
  for (int kk = 0; kk < K; kk += 32) {
    __builtin_prefetch(arow + kk + 64, 0, 0);   // global_prefetch_b8
    Frag16 a;
#pragma unroll
    for (int p = 0; p < 8; ++p) {
      int k = kk + half * 16 + 2 * p;
      a.u[p] = *(const unsigned*)(arow + k);    // (k, k+1) pair
    }
    Frag16 bfr;
    if (bTrans) {
#pragma unroll
      for (int p = 0; p < 8; ++p) {
        int k = kk + half * 16 + 2 * p;
        bfr.u[p] = *(const unsigned*)(brow + k);
      }
    } else {
#pragma unroll
      for (int p = 0; p < 8; ++p) {
        int k = kk + half * 16 + 2 * p;
        unsigned lo = *(const unsigned short*)(Bb + (size_t)k * ldb + ncol);
        unsigned hi = *(const unsigned short*)(Bb + (size_t)(k + 1) * ldb + ncol);
        bfr.u[p] = lo | (hi << 16);
      }
    }
    acc = __builtin_amdgcn_wmma_f32_16x16x32_bf16(false, a.v, false, bfr.v,
                                                  (short)0, acc, false, false);
  }
#pragma unroll
  for (int r = 0; r < 8; ++r) {
    int m = tm * 16 + r + 8 * half;
    Db[(size_t)m * ldo + ncol] = (OutT)acc[r];
  }
}

// ---------------------------------------------------------------------------
// Fused attention.
//  LDS: S   f32 [32][1024]  (scores)          128 KB @ 0
//       P   bf16[32][1024]  (attn, bf16)       64 KB @ 131072
//       Tb  bf16[256][32]   (v_c tile, TDM)    16 KB @ 196608
// Block = 64 threads (2 waves), each wave owns 16 query rows.
// ---------------------------------------------------------------------------
#define LDS_P_OFF  (32 * DIMNK * 4)
#define LDS_TB_OFF (LDS_P_OFF + 32 * DIMNK * 2)
#define LDS_TOTAL  (LDS_TB_OFF + DIMC * 32 * 2)

__global__ void attn_kernel(const __bf16* __restrict__ q,     // [B, N, 32]
                            const __bf16* __restrict__ kt,    // [B, Nk, 32]
                            const __bf16* __restrict__ vc,    // [B, C, Nk]
                            __bf16* __restrict__ out_t) {     // [B, N, C]
  extern __shared__ char smem[];
  float*  S  = (float*)smem;
  __bf16* P  = (__bf16*)(smem + LDS_P_OFF);
  __bf16* Tb = (__bf16*)(smem + LDS_TB_OFF);

  const int b    = blockIdx.y;
  const int wave = threadIdx.x >> 5;
  const int lane = threadIdx.x & 31;
  const int half = lane >> 4;
  const int l16  = lane & 15;
  const int qrow0 = blockIdx.x * 32 + wave * 16;

  const __bf16* qb = q  + (size_t)b * DIMN * DIMD;
  const __bf16* kb = kt + (size_t)b * DIMNK * DIMD;
  const __bf16* vb = vc + (size_t)b * DIMC * DIMNK;
  float*  Sw = S + (size_t)wave * 16 * DIMNK;
  __bf16* Pw = P + (size_t)wave * 16 * DIMNK;

  // --- phase 1: S = (q @ k^T) / sqrt(d)
  Frag16 afrag;
  {
    const __bf16* arow = qb + (size_t)(qrow0 + l16) * DIMD;
#pragma unroll
    for (int p = 0; p < 8; ++p) {
      int k = half * 16 + 2 * p;
      afrag.u[p] = *(const unsigned*)(arow + k);
    }
  }
  const float scale = 0.17677669529663687f;     // 1/sqrt(32)
  for (int ct = 0; ct < DIMNK / 16; ++ct) {
    // B(k=j, n=key m): k_t row m has contiguous j
    Frag16 bf;
    const __bf16* krow = kb + (size_t)(ct * 16 + l16) * DIMD;
#pragma unroll
    for (int p = 0; p < 8; ++p) {
      bf.u[p] = *(const unsigned*)(krow + half * 16 + 2 * p);
    }
    v8f acc = {};
    acc = __builtin_amdgcn_wmma_f32_16x16x32_bf16(false, afrag.v, false, bf.v,
                                                  (short)0, acc, false, false);
#pragma unroll
    for (int r = 0; r < 8; ++r) {
      int m = r + 8 * half;
      Sw[(size_t)m * DIMNK + ct * 16 + l16] = acc[r] * scale;
    }
  }
  __syncthreads();

  // --- phase 2: softmax rows -> bf16 attn in P
  for (int r = 0; r < 16; ++r) {
    float* row = Sw + (size_t)r * DIMNK;
    __bf16* prow = Pw + (size_t)r * DIMNK;
    float mx = -3.4e38f;
    for (int i = lane; i < DIMNK; i += 32) mx = fmaxf(mx, row[i]);
    for (int off = 16; off > 0; off >>= 1) mx = fmaxf(mx, __shfl_xor(mx, off, 32));
    float sum = 0.f;
    for (int i = lane; i < DIMNK; i += 32) {
      float e = __expf(row[i] - mx);
      row[i] = e;
      sum += e;
    }
    for (int off = 16; off > 0; off >>= 1) sum += __shfl_xor(sum, off, 32);
    float inv = 1.0f / sum;
    for (int i = lane; i < DIMNK; i += 32) prow[i] = (__bf16)(row[i] * inv);
  }
  __syncthreads();

  // --- phase 3: out[16 x 256] = attn[16 x Nk] @ v_c^T, 16 resident accums
  v8f acc[16];
#pragma unroll
  for (int ct = 0; ct < 16; ++ct) acc[ct] = (v8f){};

  for (int ks = 0; ks < DIMNK / 32; ++ks) {
    // stage v_c tile [256 ch rows x 32 keys] into Tb
#if USE_TDM
    if (threadIdx.x < 32) {           // wave 0 issues one TDM op
      unsigned long long ga = (unsigned long long)(size_t)vb
                            + (unsigned long long)ks * 32 * 2;
      v4u g0;
      g0[0] = 1u;                                        // count=1 (valid D#)
      g0[1] = (unsigned)LDS_TB_OFF;                      // lds_addr (dyn LDS @0)
      g0[2] = (unsigned)(ga & 0xffffffffu);              // global_addr lo
      g0[3] = (unsigned)((ga >> 32) & 0x1ffffffu) | (2u << 30);  // hi | type=2
      v8i g1;
      g1[0] = 1 << 16;                                   // data_size=2B, mask=0
      g1[1] = (int)((DIMNK & 0xffff) << 16);             // tensor_dim0 lo
      g1[2] = (int)(((unsigned)DIMNK >> 16) | ((DIMC & 0xffff) << 16)); // dim0 hi | dim1 lo
      g1[3] = (int)(32u << 16);                          // dim1 hi=0 | tile_dim0=32
      g1[4] = DIMC;                                      // tile_dim1=256, tile_dim2=0
      g1[5] = DIMNK;                                     // dim0_stride lo
      g1[6] = 0;                                         // stride hi | dim1_stride lo
      g1[7] = 0;
      v4i z4 = {};
      v8i z8 = {};
      // 6-arg (clang-23 / therock) form: groups 2,3 unused (2D tensor)
      __builtin_amdgcn_tensor_load_to_lds(g0, g1, z4, z4, z8, 0);
      __builtin_amdgcn_s_wait_tensorcnt(0);
    }
#else
    for (int i = threadIdx.x; i < DIMC * 16; i += 64) {  // dword copy fallback
      int c = i >> 4, j = i & 15;
      ((unsigned*)Tb)[i] =
          *(const unsigned*)(vb + (size_t)c * DIMNK + ks * 32 + 2 * j);
    }
#endif
    __syncthreads();

    // A fragment from bf16 attn in LDS (dword pairs)
    Frag16 aa;
    const __bf16* prow = Pw + (size_t)l16 * DIMNK + ks * 32;
#pragma unroll
    for (int p = 0; p < 8; ++p) {
      int k = half * 16 + 2 * p;
      aa.u[p] = *(const unsigned*)(prow + k);
    }
#pragma unroll
    for (int ct = 0; ct < 16; ++ct) {
      Frag16 bb;
      const __bf16* trow = Tb + (size_t)(ct * 16 + l16) * 32;
#pragma unroll
      for (int p = 0; p < 8; ++p) {
        bb.u[p] = *(const unsigned*)(trow + half * 16 + 2 * p);
      }
      acc[ct] = __builtin_amdgcn_wmma_f32_16x16x32_bf16(false, aa.v, false, bb.v,
                                                        (short)0, acc[ct],
                                                        false, false);
    }
    __syncthreads();   // protect Tb before next stage
  }

  __bf16* ob = out_t + (size_t)b * DIMN * DIMC;
#pragma unroll
  for (int ct = 0; ct < 16; ++ct) {
#pragma unroll
    for (int r = 0; r < 8; ++r) {
      int m = r + 8 * half;
      ob[(size_t)(qrow0 + m) * DIMC + ct * 16 + l16] = (__bf16)acc[ct][r];
    }
  }
}

// ---------------------------------------------------------------------------
extern "C" void kernel_launch(void* const* d_in, const int* in_sizes, int n_in,
                              void* d_out, int out_size, void* d_ws, size_t ws_size,
                              hipStream_t stream) {
  const float* x     = (const float*)d_in[0];
  const float* wq    = (const float*)d_in[1];
  const float* wk    = (const float*)d_in[2];
  const float* wv    = (const float*)d_in[3];
  const float* wo    = (const float*)d_in[4];
  const float* gamma = (const float*)d_in[5];
  float* out = (float*)d_out;

  char* ws = (char*)d_ws;
  size_t off = 0;
  auto alloc = [&](size_t bytes) {
    char* p = ws + off;
    off += (bytes + 255) & ~(size_t)255;
    return p;
  };
  __bf16* wq_bf  = (__bf16*)alloc((size_t)DIMD * DIMC * 2);
  __bf16* wk_bf  = (__bf16*)alloc((size_t)DIMD * DIMC * 2);
  __bf16* wv_bf  = (__bf16*)alloc((size_t)DIMC * DIMC * 2);
  __bf16* wo_bf  = (__bf16*)alloc((size_t)DIMC * DIMC * 2);
  __bf16* x_t    = (__bf16*)alloc((size_t)DIMB * DIMN * DIMC * 2);
  __bf16* xkv_c  = (__bf16*)alloc((size_t)DIMB * DIMC * DIMNK * 2);
  __bf16* xkv_t  = (__bf16*)alloc((size_t)DIMB * DIMNK * DIMC * 2);
  __bf16* qmat   = (__bf16*)alloc((size_t)DIMB * DIMN * DIMD * 2);
  __bf16* kmat_t = (__bf16*)alloc((size_t)DIMB * DIMNK * DIMD * 2);
  __bf16* v_c    = (__bf16*)alloc((size_t)DIMB * DIMC * DIMNK * 2);
  __bf16* out_t  = (__bf16*)alloc((size_t)DIMB * DIMN * DIMC * 2);
  float*  y_t    = (float*) alloc((size_t)DIMB * DIMN * DIMC * 4);

  cvt_f32_bf16_kernel<<<(DIMD * DIMC + 255) / 256, 256, 0, stream>>>(wq, wq_bf, DIMD * DIMC);
  cvt_f32_bf16_kernel<<<(DIMD * DIMC + 255) / 256, 256, 0, stream>>>(wk, wk_bf, DIMD * DIMC);
  cvt_f32_bf16_kernel<<<(DIMC * DIMC + 255) / 256, 256, 0, stream>>>(wv, wv_bf, DIMC * DIMC);
  cvt_f32_bf16_kernel<<<(DIMC * DIMC + 255) / 256, 256, 0, stream>>>(wo, wo_bf, DIMC * DIMC);

  transpose_x_kernel<<<(DIMB * DIMN * DIMC + 255) / 256, 256, 0, stream>>>(x, x_t);
  avgpool_kernel<<<(DIMB * DIMC * DIMNK + 255) / 256, 256, 0, stream>>>(x, xkv_c, xkv_t);

  // q[B,N,32] = x_t @ wq^T
  wmma_gemm_kernel<__bf16><<<(256 * 2 * 8) / 4, 128, 0, stream>>>(
      x_t, (long long)DIMN * DIMC, DIMC,
      wq_bf, 0, DIMC, 1,
      qmat, (long long)DIMN * DIMD, DIMD,
      256, 2, DIMB, DIMC);

  // k_t[B,Nk,32] = xkv_t @ wk^T
  wmma_gemm_kernel<__bf16><<<(64 * 2 * 8) / 4, 128, 0, stream>>>(
      xkv_t, (long long)DIMNK * DIMC, DIMC,
      wk_bf, 0, DIMC, 1,
      kmat_t, (long long)DIMNK * DIMD, DIMD,
      64, 2, DIMB, DIMC);

  // v_c[B,C,Nk] = wv @ xkv_c
  wmma_gemm_kernel<__bf16><<<(16 * 64 * 8) / 4, 128, 0, stream>>>(
      wv_bf, 0, DIMC,
      xkv_c, (long long)DIMC * DIMNK, DIMNK, 0,
      v_c, (long long)DIMC * DIMNK, DIMNK,
      16, 64, DIMB, DIMC);

  // fused attention
  {
    dim3 grid(DIMN / 32, DIMB);
    attn_kernel<<<grid, 64, LDS_TOTAL, stream>>>(qmat, kmat_t, v_c, out_t);
  }

  // y_t[B,N,C] = out_t @ wo^T (f32)
  wmma_gemm_kernel<float><<<(256 * 16 * 8) / 4, 128, 0, stream>>>(
      out_t, (long long)DIMN * DIMC, DIMC,
      wo_bf, 0, DIMC, 1,
      y_t, (long long)DIMN * DIMC, DIMC,
      256, 16, DIMB, DIMC);

  final_add_kernel<<<(DIMB * DIMC * DIMN + 255) / 256, 256, 0, stream>>>(
      x, y_t, gamma, out);
}
